// MixedFFN_15410342658373
// MI455X (gfx1250) — compile-verified, matrix operand order checked by reading, not run on hardware
//
#include <hip/hip_runtime.h>

typedef __attribute__((ext_vector_type(16))) __bf16 v16bf;
typedef __attribute__((ext_vector_type(2)))  __bf16 v2bf;
typedef __attribute__((ext_vector_type(8)))  float  v8f;

#define HIDDEN  1024
#define FFN     4096
#define NSEXP   16
#define NBATCH  8
#define SEQLEN  4096
#define NTOK    (NBATCH * SEQLEN)   // 32768 dense tokens
#define NTILE   (NTOK / 16)         // 2048 dense 16-row tiles

// LDS strides (bf16 elements); multiples of 16 elems = 32 B so v16bf loads align.
#define XS_LD   1040                // 1024 + 16 pad
#define HCHUNK  2048                // FFN processed in 2 chunks of 2048
#define H_LD    2064                // 2048 + 16 pad
#define KSTEP   64                  // K depth per staged tile (2 WMMA)
#define BS_LD   80                  // KSTEP + 16 pad (bf16 elems, col-major)
#define BS_LD32 (BS_LD / 2)         // stride in packed uint32

// Packs to a single v_cvt_pk_bf16_f32 (lo -> [15:0], hi -> [31:16]).
__device__ __forceinline__ unsigned pk2(float lo, float hi) {
    v2bf p = {(__bf16)lo, (__bf16)hi};
    return __builtin_bit_cast(unsigned, p);
}

// Stage a KSTEP x 128 fp32 weight tile -> col-major bf16 LDS tile.
// 32-bit element offsets keep the weight base in SGPRs (saddr+voffset form).
__device__ __forceinline__ void stage_w(const float* __restrict__ W, unsigned ld,
                                        unsigned kbase, unsigned nbase,
                                        unsigned* __restrict__ bs32) {
    const unsigned t = threadIdx.x;
    #pragma unroll
    for (int i = 0; i < 4; ++i) {
        unsigned idx = t + (i << 8);          // 0..1023
        unsigned r2  = idx >> 5;              // row pair 0..31 -> rows 2r2, 2r2+1
        unsigned c4  = (idx & 31) << 2;       // col 0,4,..,124
        unsigned off = (kbase + 2 * r2) * ld + nbase + c4;
        const float4 a = *(const float4*)(W + off);
        const float4 b = *(const float4*)(W + off + ld);
        unsigned* p = bs32 + c4 * BS_LD32 + r2;     // packed K pair at [col][2r2]
        p[0 * BS_LD32] = pk2(a.x, b.x);
        p[1 * BS_LD32] = pk2(a.y, b.y);
        p[2 * BS_LD32] = pk2(a.z, b.z);
        p[3 * BS_LD32] = pk2(a.w, b.w);
    }
}

__global__ __launch_bounds__(256) void mixed_ffn_fused(
    const float* __restrict__ seq_x, const int* __restrict__ seq_mask,
    const float* __restrict__ ns_x,  const int* __restrict__ ns_mask,
    const float* __restrict__ sup_w, const float* __restrict__ sup_b,
    const float* __restrict__ sdn_w, const float* __restrict__ sdn_b,
    const float* __restrict__ nup_w, const float* __restrict__ nup_b,
    const float* __restrict__ ndn_w, const float* __restrict__ ndn_b,
    float* __restrict__ out)
{
    __shared__ __align__(64) __bf16   Xs[16 * XS_LD];        //  33.3 KB
    __shared__ __align__(64) __bf16   Hs[16 * H_LD];         //  66.0 KB
    __shared__ __align__(64) unsigned Bs[2][128 * BS_LD32];  //  40.0 KB ping-pong

    const int t    = threadIdx.x;
    const int wave = t >> 5;
    const int lane = t & 31;
    const int lrow = lane & 15;          // A-row / B-col inside 16x16 tile
    const int koff = (lane >> 4) * 16;   // K sub-offset (lanes 16-31: K+16)
    const int mhi  = 8 * (lane >> 4);    // C/D: lanes 16-31 hold M+8

    const int  tile  = blockIdx.x;
    const bool dense = (tile < NTILE);
    const int  ex    = tile - NTILE;     // expert index for grouped path

    const float* Wup = dense ? sup_w : (nup_w + (size_t)ex * HIDDEN * FFN);
    const float* Bup = dense ? sup_b : (nup_b + (size_t)ex * FFN);
    const float* Wdn = dense ? sdn_w : (ndn_w + (size_t)ex * FFN * HIDDEN);
    const float* Bdn = dense ? sdn_b : (ndn_b + (size_t)ex * HIDDEN);

    // ---- stage X tile (16 x 1024) fp32 -> bf16 LDS, float4 in / b64 out ----
    #pragma unroll
    for (int i = 0; i < 16; ++i) {
        unsigned idx = (unsigned)t + (i << 8);   // 0..4095 float4 slots
        unsigned row = idx >> 8;                 // 0..15
        unsigned c4  = (idx & 255) << 2;         // 0..1020
        float4 v = make_float4(0.f, 0.f, 0.f, 0.f);
        if (dense)        v = *(const float4*)(seq_x + (unsigned)(tile * 16 + row) * HIDDEN + c4);
        else if (row < 8) v = *(const float4*)(ns_x + (row * NSEXP + ex) * HIDDEN + c4);
        *(uint2*)(Xs + row * XS_LD + c4) = make_uint2(pk2(v.x, v.y), pk2(v.z, v.w));
    }
    __syncthreads();

    v8f accY[8];                         // persistent Y accumulators (8 col-blocks)
    #pragma unroll
    for (int cb = 0; cb < 8; ++cb)
        accY[cb] = (v8f){0.f, 0.f, 0.f, 0.f, 0.f, 0.f, 0.f, 0.f};

    const int boff = (wave * 16 + lrow) * BS_LD + koff;   // B-operand LDS offset

    for (int chunk = 0; chunk < 2; ++chunk) {
        const int fbase = chunk * HCHUNK;

        // ===== phase 1: Hs = silu(X * Wup[:, fbase:fbase+2048] + b) =====
        for (int nb = 0; nb < 16; ++nb) {          // 16 blocks of 128 FFN cols
            const int ncb = fbase + nb * 128;
            v8f acc = (v8f){0.f, 0.f, 0.f, 0.f, 0.f, 0.f, 0.f, 0.f};
            stage_w(Wup, FFN, 0, ncb, Bs[0]);      // prime pipeline (K 0..63)
            for (int ks = 0; ks < HIDDEN / KSTEP; ++ks) {       // 16 steps
                __syncthreads();                   // tile ks visible; buf ks^1 free
                if (ks + 1 < HIDDEN / KSTEP)
                    stage_w(Wup, FFN, (unsigned)(ks + 1) * KSTEP, ncb, Bs[(ks + 1) & 1]);
                if (ks + 2 < HIDDEN / KSTEP)       // reach 2 tiles ahead
                    __builtin_prefetch(Wup + (unsigned)((ks + 2) * KSTEP + (t >> 2)) * FFN
                                           + ncb + (t & 3) * 32, 0, 1);
                const __bf16* bb = (const __bf16*)Bs[ks & 1];
                #pragma unroll
                for (int kk = 0; kk < KSTEP; kk += 32) {
                    v16bf a = *(const v16bf*)(Xs + lrow * XS_LD + ks * KSTEP + kk + koff);
                    v16bf b = *(const v16bf*)(bb + boff + kk);
                    acc = __builtin_amdgcn_wmma_f32_16x16x32_bf16(
                              false, a, false, b, (short)0, acc, false, false);
                }
            }
            // bias + silu -> bf16 Hs (each wave owns its own 16 cols)
            #pragma unroll
            for (int j = 0; j < 8; ++j) {
                int m  = j + mhi;
                int nc = nb * 128 + wave * 16 + lrow;   // chunk-local col
                float h = acc[j] + Bup[fbase + nc];
                h = h / (1.0f + __expf(-h));            // silu
                Hs[m * H_LD + nc] = (__bf16)h;
            }
        }

        // ===== phase 2: Y += Hs * Wdn[fbase:fbase+2048, :] =====
        #pragma unroll                               // keep accY in VGPRs
        for (int cb = 0; cb < 8; ++cb) {             // 8 blocks of 128 out cols
            stage_w(Wdn, HIDDEN, (unsigned)fbase, cb * 128, Bs[0]);
            for (int ks = 0; ks < HCHUNK / KSTEP; ++ks) {       // 32 steps
                __syncthreads();
                if (ks + 1 < HCHUNK / KSTEP)
                    stage_w(Wdn, HIDDEN, (unsigned)(fbase + (ks + 1) * KSTEP),
                            cb * 128, Bs[(ks + 1) & 1]);
                if (ks + 2 < HCHUNK / KSTEP)
                    __builtin_prefetch(Wdn + (unsigned)(fbase + (ks + 2) * KSTEP + (t >> 2)) * HIDDEN
                                           + cb * 128 + (t & 3) * 32, 0, 1);
                const __bf16* bb = (const __bf16*)Bs[ks & 1];
                #pragma unroll
                for (int kk = 0; kk < KSTEP; kk += 32) {
                    v16bf a = *(const v16bf*)(Hs + lrow * H_LD + ks * KSTEP + kk + koff);
                    v16bf b = *(const v16bf*)(bb + boff + kk);
                    accY[cb] = __builtin_amdgcn_wmma_f32_16x16x32_bf16(
                                   false, a, false, b, (short)0, accY[cb], false, false);
                }
            }
        }
        // next chunk's phase-1 barriers keep all waves past these Hs reads
    }

    // ---- epilogue: down-bias, mask, fp32 store ----
    #pragma unroll
    for (int cb = 0; cb < 8; ++cb) {
        #pragma unroll
        for (int j = 0; j < 8; ++j) {
            int m  = j + mhi;
            unsigned nc = cb * 128 + wave * 16 + lrow;   // 0..1023
            float y = accY[cb][j] + Bdn[nc];
            if (dense) {
                unsigned tok = (unsigned)tile * 16 + m;
                y *= (float)seq_mask[tok];
                out[tok * (unsigned)HIDDEN + nc] = y;
            } else if (m < 8) {                          // rows 8..15 are padding
                y *= (float)ns_mask[m * NSEXP + ex];
                out[(unsigned)NTOK * HIDDEN + (unsigned)(m * NSEXP + ex) * HIDDEN + nc] = y;
            }
        }
    }
}

extern "C" void kernel_launch(void* const* d_in, const int* in_sizes, int n_in,
                              void* d_out, int out_size, void* d_ws, size_t ws_size,
                              hipStream_t stream) {
    (void)in_sizes; (void)n_in; (void)d_ws; (void)ws_size; (void)out_size;
    const float* seq_x  = (const float*)d_in[0];
    const int*   seq_m  = (const int*)  d_in[1];
    const float* ns_x   = (const float*)d_in[2];
    const int*   ns_m   = (const int*)  d_in[3];
    const float* sup_w  = (const float*)d_in[4];
    const float* sup_b  = (const float*)d_in[5];
    const float* sdn_w  = (const float*)d_in[6];
    const float* sdn_b  = (const float*)d_in[7];
    const float* nup_w  = (const float*)d_in[8];
    const float* nup_b  = (const float*)d_in[9];
    const float* ndn_w  = (const float*)d_in[10];
    const float* ndn_b  = (const float*)d_in[11];
    float* out = (float*)d_out;

    hipLaunchKernelGGL(mixed_ffn_fused, dim3(NTILE + NSEXP), dim3(256), 0, stream,
                       seq_x, seq_m, ns_x, ns_m,
                       sup_w, sup_b, sdn_w, sdn_b,
                       nup_w, nup_b, ndn_w, ndn_b, out);
}
